// WindowAttention3D_49254684951130
// MI455X (gfx1250) — matrix operand order repaired and checked by reading, not compiled, not run on hardware
//
#include <hip/hip_runtime.h>
#include <hip/hip_bf16.h>

typedef __attribute__((ext_vector_type(16))) _Float16 v16h;
typedef __attribute__((ext_vector_type(4)))  _Float16 v4h;
typedef __attribute__((ext_vector_type(8)))  float    v8f;
typedef __attribute__((ext_vector_type(4)))  unsigned int u32x4;
typedef __attribute__((ext_vector_type(8)))  int      i32x8;
typedef __attribute__((ext_vector_type(4)))  int      i32x4;

#define NUM_HEADS 12
#define HD 32
#define NTOK 64
#define CDIM 384

#if defined(__gfx1250__) && __has_builtin(__builtin_amdgcn_tensor_load_to_lds)
#define USE_TDM 1
#else
#define USE_TDM 0
#endif

// ---------------------------------------------------------------------------
// WMMA helpers (v_wmma_f32_16x16x32_f16, wave32)
// ---------------------------------------------------------------------------
__device__ __forceinline__ v8f wmma_f16(v16h a, v16h b, v8f c) {
    return __builtin_amdgcn_wmma_f32_16x16x32_f16(
        false, a, false, b, (short)0, c, false, false);
}

// A-matrix 16x32 fragment (row-major LDS). Lane 0-15: M=mo+l, K=j;
// lanes 16-31: K=j+16.
__device__ __forceinline__ v16h load_a_frag(const _Float16* As, int lda,
                                            int mo, int ko, int lane) {
    v16h a;
    const int row = mo + (lane & 15);
    const int hi  = lane >> 4;
#pragma unroll
    for (int j = 0; j < 16; ++j)
        a[j] = As[row * lda + (ko + j + 16 * hi)];
    return a;
}

// B-matrix 32x16 fragment from row-major (K x N) LDS. Lane 0-15: N=no+l, K=j;
// lanes 16-31: K=j+16.
__device__ __forceinline__ v16h load_b_frag(const _Float16* Bs, int ldb,
                                            int ko, int no, int lane) {
    v16h b;
    const int col = no + (lane & 15);
    const int hi  = lane >> 4;
#pragma unroll
    for (int j = 0; j < 16; ++j)
        b[j] = Bs[(ko + j + 16 * hi) * ldb + col];
    return b;
}

// B fragment where logical B is transpose of a row-major (N x K) LDS tile.
__device__ __forceinline__ v16h load_bt_frag(const _Float16* Kt, int ldk,
                                             int ko, int no, int lane) {
    v16h b;
    const int col = no + (lane & 15);
    const int hi  = lane >> 4;
#pragma unroll
    for (int j = 0; j < 16; ++j)
        b[j] = Kt[col * ldk + (ko + j + 16 * hi)];
    return b;
}

// ---------------------------------------------------------------------------
// Tensor Data Mover: 2D f16 tile (tile_w x tile_h) from row-major global
// (row stride = stride_elems) into LDS at byte offset lds_off, with LDS row
// padding of 1 DWORD every 2^(pad_code+1) DWORDs.
// D# layout per CDNA5 ISA ch.8 (group0: count/lds/global/type, group1: dims).
// Toolchain uses the 6-arg builtin: (g0, g1, g2, g3, g4, cpol).
// ---------------------------------------------------------------------------
#if USE_TDM
__device__ __forceinline__ void tdm_load_2d(unsigned int lds_off,
                                            const _Float16* gsrc,
                                            unsigned int stride_elems,
                                            unsigned int tile_w,
                                            unsigned int tile_h,
                                            int pad_code) {
    const unsigned long long ga = (unsigned long long)(uintptr_t)gsrc;
    u32x4 g0;
    g0[0] = 1u;                                         // count=1 (valid user D#)
    g0[1] = lds_off;                                    // lds_addr
    g0[2] = (unsigned int)(ga & 0xFFFFFFFFu);           // global_addr[31:0]
    g0[3] = (unsigned int)((ga >> 32) & 0x01FFFFFFu)    // global_addr[56:32]
          | 0x80000000u;                                // type=2 ("image")
    i32x8 g1;
    g1[0] = (1 << 16)                                   // data_size = 2 bytes
          | (1 << 20)                                   // pad_enable
          | (pad_code << 22)                            // pad_interval
          | (0 << 25);                                  // pad_amount = 1 DWORD
    const unsigned int td = 0x40000000u;                // huge tensor dims: no OOB
    g1[1] = (int)((td & 0xFFFFu) << 16);                // tensor_dim0[15:0]
    g1[2] = (int)((td >> 16) | ((td & 0xFFFFu) << 16)); // tdim0[31:16] | tdim1[15:0]
    g1[3] = (int)(((td >> 16) & 0xFFFFu)                // tensor_dim1[31:16]
          | (tile_w << 16));                            // tile_dim0
    g1[4] = (int)(tile_h & 0xFFFFu);                    // tile_dim1 (tile_dim2=0)
    g1[5] = (int)stride_elems;                          // tensor_dim0_stride[31:0]
    g1[6] = 0;                                          // stride hi / dim1_stride lo
    g1[7] = 0;
    const i32x4 z4 = {0, 0, 0, 0};
    const i32x8 z8 = {0, 0, 0, 0, 0, 0, 0, 0};
    __builtin_amdgcn_tensor_load_to_lds(g0, g1, z4, z4, z8, 0);
}
#endif

// ---------------------------------------------------------------------------
// Kernel 0: f32 -> f16 bulk convert (vectorized, grid-stride)
// ---------------------------------------------------------------------------
__global__ __launch_bounds__(256)
void wa3d_cvt_f16(const float* __restrict__ s, _Float16* __restrict__ d, long n4) {
    const float4* s4 = (const float4*)s;
    v4h* d4 = (v4h*)d;
    for (long i = blockIdx.x * 256L + threadIdx.x; i < n4;
         i += (long)gridDim.x * 256L) {
        const float4 v = s4[i];
        v4h h;
        h[0] = (_Float16)v.x; h[1] = (_Float16)v.y;
        h[2] = (_Float16)v.z; h[3] = (_Float16)v.w;
        d4[i] = h;
    }
}

// ---------------------------------------------------------------------------
// Kernel 1: relative-position-bias MLP -> bias[12][64][64] f32
// ---------------------------------------------------------------------------
__global__ __launch_bounds__(64)
void wa3d_bias_mlp(const float* __restrict__ w1, const float* __restrict__ b1,
                   const float* __restrict__ w2, const float* __restrict__ b2,
                   float* __restrict__ bias) {
    const int i = blockIdx.x, j = threadIdx.x;
    const float r0 = (float)((i >> 4)       - (j >> 4));
    const float r1 = (float)(((i >> 2) & 3) - ((j >> 2) & 3));
    const float r2 = (float)((i & 3)        - (j & 3));
    float acc[NUM_HEADS];
#pragma unroll
    for (int h = 0; h < NUM_HEADS; ++h) acc[h] = b2[h];
    for (int u = 0; u < 64; ++u) {
        const float hv =
            fmaxf(r0 * w1[u] + r1 * w1[64 + u] + r2 * w1[128 + u] + b1[u], 0.f);
#pragma unroll
        for (int h = 0; h < NUM_HEADS; ++h) acc[h] += hv * w2[u * NUM_HEADS + h];
    }
#pragma unroll
    for (int h = 0; h < NUM_HEADS; ++h)
        bias[((h * 64 + i) * 64) + j] = acc[h];
}

// ---------------------------------------------------------------------------
// Kernel 2/4: WMMA GEMM  D[M,N] = A[M,K]h * B[K,N]h + bias[N]
//   256 threads = 8 waves; block tile 128x64; wave tile 32x32 (4 wmma/K-step).
//   Double-buffered LDS tiles filled by TDM (wave 0) when available.
// ---------------------------------------------------------------------------
struct alignas(16) GemmLds {
    _Float16 As[2][128 * 34];   // 2 x 8704 B (A rows padded 32->34 halves)
    _Float16 Bs[2][32 * 66];    // 2 x 4224 B (B rows padded 64->66 halves)
};

template <bool OUT_HALF>
__global__ __launch_bounds__(256)
void wa3d_gemm_wmma(const _Float16* __restrict__ A, const _Float16* __restrict__ B,
                    const float* __restrict__ bias, void* __restrict__ D,
                    int M, int N, int K) {
    __shared__ GemmLds lds;

    const int tid  = threadIdx.x;
    const int w    = tid >> 5;
    const int lane = tid & 31;
    const int bm   = blockIdx.y * 128;
    const int bn   = blockIdx.x * 64;
    const int mo   = (w >> 1) * 32;        // wave M origin within block tile
    const int no   = (w & 1) * 32;         // wave N origin

    const int S = K / 32;                  // K stages

#if USE_TDM
    if (w == 0) {
        tdm_load_2d((unsigned)__builtin_offsetof(GemmLds, As),
                    A + (size_t)bm * K, K, 32, 128, /*16 DW rows*/ 3);
        tdm_load_2d((unsigned)__builtin_offsetof(GemmLds, Bs),
                    B + (size_t)bn, N, 64, 32, /*32 DW rows*/ 4);
    }
#else
    for (int i = tid; i < 128 * 32; i += 256) {
        const int r = i >> 5, c = i & 31;
        lds.As[0][r * 34 + c] = A[(size_t)(bm + r) * K + c];
    }
    for (int i = tid; i < 32 * 64; i += 256) {
        const int r = i >> 6, c = i & 63;
        lds.Bs[0][r * 66 + c] = B[(size_t)r * N + (bn + c)];
    }
#endif

    v8f c00 = {}, c01 = {}, c10 = {}, c11 = {};

    for (int s = 0; s < S; ++s) {
        const int nb = (s + 1) & 1;
        if (s + 1 < S) {
            const int k1 = (s + 1) * 32;
#if USE_TDM
            if (w == 0) {
                tdm_load_2d((unsigned)(__builtin_offsetof(GemmLds, As) + nb * 128 * 34 * 2),
                            A + (size_t)bm * K + k1, K, 32, 128, 3);
                tdm_load_2d((unsigned)(__builtin_offsetof(GemmLds, Bs) + nb * 32 * 66 * 2),
                            B + (size_t)k1 * N + bn, N, 64, 32, 4);
                __builtin_amdgcn_s_wait_tensorcnt(2);  // stage s resident
            }
#else
            for (int i = tid; i < 128 * 32; i += 256) {
                const int r = i >> 5, c = i & 31;
                lds.As[nb][r * 34 + c] = A[(size_t)(bm + r) * K + (k1 + c)];
            }
            for (int i = tid; i < 32 * 64; i += 256) {
                const int r = i >> 6, c = i & 63;
                lds.Bs[nb][r * 66 + c] = B[(size_t)(k1 + r) * N + (bn + c)];
            }
#endif
        }
#if USE_TDM
        else if (w == 0) {
            __builtin_amdgcn_s_wait_tensorcnt(0);      // drain final stage
        }
#endif
        __syncthreads();                               // stage s published

        const _Float16* Asb = lds.As[s & 1];
        const _Float16* Bsb = lds.Bs[s & 1];
        const v16h a0 = load_a_frag(Asb, 34, mo,      0, lane);
        const v16h a1 = load_a_frag(Asb, 34, mo + 16, 0, lane);
        const v16h b0 = load_b_frag(Bsb, 66, 0, no,      lane);
        const v16h b1 = load_b_frag(Bsb, 66, 0, no + 16, lane);
        c00 = wmma_f16(a0, b0, c00);
        c01 = wmma_f16(a0, b1, c01);
        c10 = wmma_f16(a1, b0, c10);
        c11 = wmma_f16(a1, b1, c11);
        __syncthreads();                               // done reading buffer s&1
    }

    // epilogue (C/D layout: N = lane&15, M = r + 8*(lane>>4))
    const int hi = lane >> 4;
    const int nc0 = bn + no + (lane & 15);
    const int nc1 = nc0 + 16;
    const float bv0 = bias[nc0];
    const float bv1 = bias[nc1];
#pragma unroll
    for (int r = 0; r < 8; ++r) {
        const int m0 = bm + mo + r + 8 * hi;
        const int m1 = m0 + 16;
        if (OUT_HALF) {
            _Float16* Dh = (_Float16*)D;
            Dh[(size_t)m0 * N + nc0] = (_Float16)(c00[r] + bv0);
            Dh[(size_t)m0 * N + nc1] = (_Float16)(c01[r] + bv1);
            Dh[(size_t)m1 * N + nc0] = (_Float16)(c10[r] + bv0);
            Dh[(size_t)m1 * N + nc1] = (_Float16)(c11[r] + bv1);
        } else {
            float* Df = (float*)D;
            Df[(size_t)m0 * N + nc0] = c00[r] + bv0;
            Df[(size_t)m0 * N + nc1] = c01[r] + bv1;
            Df[(size_t)m1 * N + nc0] = c10[r] + bv0;
            Df[(size_t)m1 * N + nc1] = c11[r] + bv1;
        }
    }
}

// ---------------------------------------------------------------------------
// Kernel 3: per-(window, head) attention, 128 threads = 4 waves.
//   S = (Q K^T) * scale + bias ; P = softmax(S) ; O = P V
// ---------------------------------------------------------------------------
struct alignas(16) AttnLds {
    _Float16 Qs[64 * 34];   // 4352 B each, rows padded 32->34
    _Float16 Ks[64 * 34];
    _Float16 Vs[64 * 34];
    float    Ss[64 * 64];   // 16384 B
    _Float16 Ps[64 * 66];   // 8448 B
};

__global__ __launch_bounds__(128)
void wa3d_attention(const _Float16* __restrict__ qkv,
                    const float* __restrict__ bias,
                    _Float16* __restrict__ attn_out) {
    __shared__ AttnLds lds;

    const int b = blockIdx.y;
    const int h = blockIdx.x;
    const int tid = threadIdx.x;
    const int w = tid >> 5, lane = tid & 31;

    const size_t base = (size_t)b * NTOK * (3 * CDIM) + (size_t)h * HD;
#if USE_TDM
    if (w == 0) {
        tdm_load_2d((unsigned)__builtin_offsetof(AttnLds, Qs), qkv + base,
                    3 * CDIM, 32, 64, 3);
        tdm_load_2d((unsigned)__builtin_offsetof(AttnLds, Ks), qkv + base + CDIM,
                    3 * CDIM, 32, 64, 3);
        tdm_load_2d((unsigned)__builtin_offsetof(AttnLds, Vs), qkv + base + 2 * CDIM,
                    3 * CDIM, 32, 64, 3);
        __builtin_amdgcn_s_wait_tensorcnt(0);
    }
#else
    for (int i = tid; i < NTOK * HD; i += 128) {
        const int n = i >> 5, d = i & 31;
        const size_t off = base + (size_t)n * (3 * CDIM) + d;
        lds.Qs[n * 34 + d] = qkv[off];
        lds.Ks[n * 34 + d] = qkv[off + CDIM];
        lds.Vs[n * 34 + d] = qkv[off + 2 * CDIM];
    }
#endif
    __syncthreads();

    const int mo = w * 16;                    // wave w owns query rows [16w,16w+16)
    const int hi = lane >> 4;
    const float scale = 0.17677669529663687f; // 1/sqrt(32)

    // ---- S = Q K^T (full K=32 in one wmma per 16x16 tile) ----
    const v16h aq = load_a_frag(lds.Qs, 34, mo, 0, lane);
#pragma unroll
    for (int nb = 0; nb < 4; ++nb) {
        const int no = nb * 16;
        const v16h bk = load_bt_frag(lds.Ks, 34, 0, no, lane);
        v8f s = {};
        s = wmma_f16(aq, bk, s);
        const int n = no + (lane & 15);
#pragma unroll
        for (int r = 0; r < 8; ++r) {
            const int m = mo + r + 8 * hi;
            lds.Ss[m * 64 + n] = s[r] * scale + bias[(h * 64 + m) * 64 + n];
        }
    }
    __syncthreads();

    // ---- row softmax (thread per query row) ----
    if (tid < 64) {
        float mx = -3.0e38f;
#pragma unroll 8
        for (int j = 0; j < 64; ++j) mx = fmaxf(mx, lds.Ss[tid * 64 + j]);
        float sum = 0.f;
#pragma unroll 8
        for (int j = 0; j < 64; ++j) {
            const float e = __expf(lds.Ss[tid * 64 + j] - mx);
            lds.Ss[tid * 64 + j] = e;
            sum += e;
        }
        const float inv = 1.f / sum;
#pragma unroll 8
        for (int j = 0; j < 64; ++j)
            lds.Ps[tid * 66 + j] = (_Float16)(lds.Ss[tid * 64 + j] * inv);
    }
    __syncthreads();

    // ---- O = P V (K=64 -> two wmma per 16x16 tile) ----
#pragma unroll
    for (int ni = 0; ni < 2; ++ni) {
        const int no = ni * 16;
        v8f o = {};
#pragma unroll
        for (int k0 = 0; k0 < 64; k0 += 32) {
            const v16h ap = load_a_frag(lds.Ps, 66, mo, k0, lane);
            const v16h bv = load_b_frag(lds.Vs, 34, k0, no, lane);
            o = wmma_f16(ap, bv, o);
        }
        const int n = no + (lane & 15);
#pragma unroll
        for (int r = 0; r < 8; ++r) {
            const int m = mo + r + 8 * hi;
            attn_out[(size_t)(b * NTOK + m) * CDIM + h * HD + n] = (_Float16)o[r];
        }
    }
}

// ---------------------------------------------------------------------------
// Launch
// ---------------------------------------------------------------------------
extern "C" void kernel_launch(void* const* d_in, const int* in_sizes, int n_in,
                              void* d_out, int out_size, void* d_ws, size_t ws_size,
                              hipStream_t stream) {
    const float* x      = (const float*)d_in[0];   // [2048,64,384]
    const float* qkv_w  = (const float*)d_in[1];   // [384,1152]
    const float* qkv_b  = (const float*)d_in[2];   // [1152]
    const float* proj_w = (const float*)d_in[3];   // [384,384]
    const float* proj_b = (const float*)d_in[4];   // [384]
    const float* mlp_w1 = (const float*)d_in[5];
    const float* mlp_b1 = (const float*)d_in[6];
    const float* mlp_w2 = (const float*)d_in[7];
    const float* mlp_b2 = (const float*)d_in[8];
    float* out = (float*)d_out;

    const int Bwin = 2048;
    const int M    = Bwin * NTOK;                  // 131072

    // workspace layout (all 16B aligned)
    char* ws = (char*)d_ws;
    float*    bias_ws = (float*)ws;                                  // 192 KB
    const size_t OFF_XH   = (size_t)1 << 20;
    const size_t OFF_QW   = OFF_XH  + (size_t)M * CDIM * 2;          // x_h: 100 MB
    const size_t OFF_PW   = OFF_QW  + (size_t)CDIM * 3 * CDIM * 2;   // qkv_w_h
    const size_t OFF_QKV  = OFF_PW  + (size_t)CDIM * CDIM * 2;       // proj_w_h
    const size_t OFF_ATTN = OFF_QKV + (size_t)M * 3 * CDIM * 2;      // qkv: 302 MB
    _Float16* xh_ws   = (_Float16*)(ws + OFF_XH);
    _Float16* qw_ws   = (_Float16*)(ws + OFF_QW);
    _Float16* pw_ws   = (_Float16*)(ws + OFF_PW);
    _Float16* qkv_ws  = (_Float16*)(ws + OFF_QKV);
    _Float16* attn_ws = (_Float16*)(ws + OFF_ATTN);                  // 101 MB

    // 0) f32 -> f16 conversions (x, qkv_w, proj_w)
    wa3d_cvt_f16<<<4096, 256, 0, stream>>>(x, xh_ws, (long)M * CDIM / 4);
    wa3d_cvt_f16<<<432,  256, 0, stream>>>(qkv_w, qw_ws, (long)CDIM * 3 * CDIM / 4);
    wa3d_cvt_f16<<<144,  256, 0, stream>>>(proj_w, pw_ws, (long)CDIM * CDIM / 4);

    // 1) relative position bias table [12][64][64]
    wa3d_bias_mlp<<<64, 64, 0, stream>>>(mlp_w1, mlp_b1, mlp_w2, mlp_b2, bias_ws);

    // 2) qkv = x @ qkv_w + qkv_b  -> f16 [M][1152]
    wa3d_gemm_wmma<true><<<dim3((3 * CDIM) / 64, M / 128), 256, 0, stream>>>(
        xh_ws, qw_ws, qkv_b, (void*)qkv_ws, M, 3 * CDIM, CDIM);

    // 3) attention per (window, head) -> f16 [M][384]
    wa3d_attention<<<dim3(NUM_HEADS, Bwin), 128, 0, stream>>>(qkv_ws, bias_ws, attn_ws);

    // 4) out = attn @ proj_w + proj_b -> f32 [M][384]
    wa3d_gemm_wmma<false><<<dim3(CDIM / 64, M / 128), 256, 0, stream>>>(
        attn_ws, pw_ws, proj_b, (void*)out, M, CDIM, CDIM);
}
